// YuanMoeLayer_9328668967831
// MI455X (gfx1250) — matrix operand order repaired and checked by reading, not compiled, not run on hardware
//
#include <hip/hip_runtime.h>

#define HIDDEN 2048
#define FFN    2048
#define NE     8
#define NTOK   2048
#define NSLOTS (2 * NTOK)

#define TILE_M 64
#define TILE_N 32
#define TILE_K 64
#define LDK 72   // LDS row stride in shorts: 144B rows -> 16B-aligned fragments,
                 // and 36*r mod 64 banks distinct for 16-lane b128 access

typedef __attribute__((ext_vector_type(16))) __bf16 v16bf;
typedef __attribute__((ext_vector_type(8)))  float  v8f;
typedef __attribute__((ext_vector_type(4)))  unsigned int v4u;

__device__ __forceinline__ unsigned short f2bf(float f) {
  unsigned int u = __float_as_uint(f);
  u += 0x7FFFu + ((u >> 16) & 1u);   // round-to-nearest-even
  return (unsigned short)(u >> 16);
}

__device__ __forceinline__ unsigned int f2bf2(float lo, float hi) {
  return (unsigned int)f2bf(lo) | ((unsigned int)f2bf(hi) << 16);
}

// ---------------------------------------------------------------- zero init
__global__ void zero_small_kernel(int* __restrict__ counts) {
  int t = threadIdx.x;
  if (t < 16) counts[t] = 0;  // counts[0..7], cursors[8..15]
}

// ---------------------------------------------------------------- router
// one wave per token: mix = x @ w_qkv.T (24 outputs), 8x8 softmax attention,
// top-2, softmax scores, atomicAdd per-expert counts.
__global__ __launch_bounds__(256) void router_kernel(
    const float* __restrict__ x, const float* __restrict__ wqkv,
    float* __restrict__ scores, int* __restrict__ indices,
    int* __restrict__ counts) {
  int lane = threadIdx.x & 31;
  int wave = threadIdx.x >> 5;
  int s = blockIdx.x * 8 + wave;

  float acc[24];
#pragma unroll
  for (int r = 0; r < 24; ++r) acc[r] = 0.f;

  const float* xr = x + (size_t)s * HIDDEN;
  for (int h = lane; h < HIDDEN; h += 32) {
    float xv = xr[h];
#pragma unroll
    for (int r = 0; r < 24; ++r)
      acc[r] = fmaf(xv, wqkv[r * HIDDEN + h], acc[r]);
  }
#pragma unroll
  for (int off = 16; off > 0; off >>= 1) {
#pragma unroll
    for (int r = 0; r < 24; ++r)
      acc[r] += __shfl_xor(acc[r], off, 32);
  }

  if (lane == 0) {
    float logits[8];
#pragma unroll
    for (int i = 0; i < 8; ++i) {
      float qi = acc[i];
      float m = -3.4e38f;
#pragma unroll
      for (int j = 0; j < 8; ++j) m = fmaxf(m, qi * acc[8 + j]);
      float den = 0.f, num = 0.f;
#pragma unroll
      for (int j = 0; j < 8; ++j) {
        float e = __expf(qi * acc[8 + j] - m);
        den += e;
        num += e * acc[16 + j];
      }
      logits[i] = num / den;
    }
    int i0 = 0; float l0 = logits[0];
#pragma unroll
    for (int j = 1; j < 8; ++j)
      if (logits[j] > l0) { l0 = logits[j]; i0 = j; }
    int i1 = -1; float l1 = -3.4e38f;
#pragma unroll
    for (int j = 0; j < 8; ++j)
      if (j != i0 && logits[j] > l1) { l1 = logits[j]; i1 = j; }
    float e1 = __expf(l1 - l0);
    float inv = 1.f / (1.f + e1);
    scores[2 * s + 0] = inv;
    scores[2 * s + 1] = e1 * inv;
    indices[2 * s + 0] = i0;
    indices[2 * s + 1] = i1;
    atomicAdd(&counts[i0], 1);
    atomicAdd(&counts[i1], 1);
  }
}

// ---------------------------------------------------------------- scan+scatter
__global__ __launch_bounds__(256) void scan_scatter_kernel(
    const int* __restrict__ counts, int* __restrict__ cursors,
    int* __restrict__ offs, const int* __restrict__ indices,
    int* __restrict__ list_token, int* __restrict__ slot_of) {
  if (threadIdx.x == 0) {
    int run = 0;
    for (int e = 0; e < NE; ++e) { offs[e] = run; run += counts[e]; }
  }
  __syncthreads();
  for (int s = threadIdx.x; s < NTOK; s += blockDim.x) {
#pragma unroll
    for (int k = 0; k < 2; ++k) {
      int e = indices[2 * s + k];
      int pos = atomicAdd(&cursors[e], 1);
      int slot = offs[e] + pos;
      list_token[slot] = s;
      slot_of[2 * s + k] = slot;
    }
  }
}

// ---------------------------------------------------------------- fc1 + SiLU
// grouped GEMM over ragged token lists: inter[slot, n] = silu(x@Wg) * (x@Wu)
// 64x32 block tile, K=64 per stage, 8 waves as 4(M) x 2(N), b128 fragment loads.
__global__ __launch_bounds__(256) void expert_fc1_kernel(
    const float* __restrict__ x, const float* __restrict__ w1,
    const int* __restrict__ list_token, const int* __restrict__ offs,
    const int* __restrict__ counts, unsigned short* __restrict__ inter) {
  int e = blockIdx.z;
  int cnt = counts[e];
  int m0 = blockIdx.y * TILE_M;
  if (m0 >= cnt) return;
  int base = offs[e] + m0;
  int rows = cnt - m0; if (rows > TILE_M) rows = TILE_M;
  int n0 = blockIdx.x * TILE_N;                  // column within [0, FFN)
  const float* w = w1 + (size_t)e * HIDDEN * (2 * FFN);

  __shared__ int toks[TILE_M];
  __shared__ __align__(16) unsigned short As[TILE_M * LDK];
  __shared__ __align__(16) unsigned short Bg[TILE_N * LDK];
  __shared__ __align__(16) unsigned short Bu[TILE_N * LDK];

  int tid = threadIdx.x;
  int lane = tid & 31, wave = tid >> 5;
  int wm = wave >> 1, wn = wave & 1;             // 4(M) x 2(N) wave grid
  int half = lane >> 4;
  int l15 = lane & 15;

  if (tid < TILE_M) {
    int rr = (tid < rows) ? tid : (rows - 1);
    toks[tid] = list_token[base + rr];
  }
  __syncthreads();

  v8f cg = {}; v8f cu = {};
  int abase = (wm * 16 + l15) * LDK;
  int bbase = (wn * 16 + l15) * LDK;

  for (int k0 = 0; k0 < HIDDEN; k0 += TILE_K) {
#pragma unroll
    for (int i = 0; i < 8; ++i) {                 // A: 64 rows x 32 fp32-pairs -> packed dwords
      int idx = tid + i * 256;
      int r = idx >> 5, kp = idx & 31;
      const float* src = &x[(size_t)toks[r] * HIDDEN + k0 + 2 * kp];
      *(unsigned int*)&As[r * LDK + 2 * kp] = f2bf2(src[0], src[1]);
    }
#pragma unroll
    for (int i = 0; i < 8; ++i) {                 // Bg/Bu: 64(k) x 32(n), stored [n][k]
      int idx = tid + i * 256;
      int kk = idx >> 5, nn = idx & 31;
      size_t ro = (size_t)(k0 + kk) * (2 * FFN);
      Bg[nn * LDK + kk] = f2bf(w[ro + n0 + nn]);
      Bu[nn * LDK + kk] = f2bf(w[ro + FFN + n0 + nn]);
    }
    if (k0 + TILE_K < HIDDEN && tid < 2 * TILE_K) { // prefetch next weight rows
      int kk = tid & 63;
      const float* p = (tid < TILE_K)
          ? &w[(size_t)(k0 + TILE_K + kk) * (2 * FFN) + n0]
          : &w[(size_t)(k0 + TILE_K + kk) * (2 * FFN) + FFN + n0];
      __builtin_prefetch(p, 0, 3);
    }
    __syncthreads();

#pragma unroll
    for (int kc = 0; kc < TILE_K; kc += 32) {
      v16bf a, bg, bu;
      ((v4u*)&a)[0]  = *(const v4u*)&As[abase + kc + 16 * half];
      ((v4u*)&a)[1]  = *(const v4u*)&As[abase + kc + 16 * half + 8];
      ((v4u*)&bg)[0] = *(const v4u*)&Bg[bbase + kc + 16 * half];
      ((v4u*)&bg)[1] = *(const v4u*)&Bg[bbase + kc + 16 * half + 8];
      ((v4u*)&bu)[0] = *(const v4u*)&Bu[bbase + kc + 16 * half];
      ((v4u*)&bu)[1] = *(const v4u*)&Bu[bbase + kc + 16 * half + 8];
      cg = __builtin_amdgcn_wmma_f32_16x16x32_bf16(false, a, false, bg, (short)0, cg, false, false);
      cu = __builtin_amdgcn_wmma_f32_16x16x32_bf16(false, a, false, bu, (short)0, cu, false, false);
    }
    __syncthreads();
  }

  int ncol = n0 + wn * 16 + l15;
#pragma unroll
  for (int i = 0; i < 8; ++i) {
    int ml = wm * 16 + i + 8 * half;
    if (ml < rows) {
      float g = cg[i], u = cu[i];
      float val = g * __builtin_amdgcn_rcpf(1.f + __expf(-g)) * u;  // silu(g)*u
      inter[(size_t)(base + ml) * FFN + ncol] = f2bf(val);
    }
  }
}

// ---------------------------------------------------------------- fc2
// buf[slot, n] = inter[slot, :] @ w2[e]   (rows are contiguous slots, A already bf16)
__global__ __launch_bounds__(256) void expert_fc2_kernel(
    const unsigned short* __restrict__ inter, const float* __restrict__ w2,
    const int* __restrict__ offs, const int* __restrict__ counts,
    float* __restrict__ buf) {
  int e = blockIdx.z;
  int cnt = counts[e];
  int m0 = blockIdx.y * TILE_M;
  if (m0 >= cnt) return;
  int base = offs[e] + m0;
  int rows = cnt - m0; if (rows > TILE_M) rows = TILE_M;
  int n0 = blockIdx.x * TILE_N;
  const float* w = w2 + (size_t)e * FFN * HIDDEN;

  __shared__ __align__(16) unsigned short As[TILE_M * LDK];
  __shared__ __align__(16) unsigned short Bs[TILE_N * LDK];

  int tid = threadIdx.x;
  int lane = tid & 31, wave = tid >> 5;
  int wm = wave >> 1, wn = wave & 1;
  int half = lane >> 4;
  int l15 = lane & 15;

  v8f c = {};
  int abase = (wm * 16 + l15) * LDK;
  int bbase = (wn * 16 + l15) * LDK;

  for (int k0 = 0; k0 < FFN; k0 += TILE_K) {
#pragma unroll
    for (int i = 0; i < 2; ++i) {                 // A: 64 rows x 128B (b128 both sides)
      int idx = tid + i * 256;                    // 0..511 16B-chunks
      int r = idx >> 3, cchunk = idx & 7;
      int rr = (r < rows) ? r : (rows - 1);
      v4u dv = *(const v4u*)&inter[(size_t)(base + rr) * FFN + k0 + 8 * cchunk];
      *(v4u*)&As[r * LDK + 8 * cchunk] = dv;
    }
#pragma unroll
    for (int i = 0; i < 8; ++i) {                 // B: 64(k) x 32(n) fp32->bf16, stored [n][k]
      int idx = tid + i * 256;
      int kk = idx >> 5, nn = idx & 31;
      Bs[nn * LDK + kk] = f2bf(w[(size_t)(k0 + kk) * HIDDEN + n0 + nn]);
    }
    if (k0 + TILE_K < FFN && tid < TILE_K) {      // prefetch next weight rows
      __builtin_prefetch(&w[(size_t)(k0 + TILE_K + tid) * HIDDEN + n0], 0, 3);
    }
    __syncthreads();

#pragma unroll
    for (int kc = 0; kc < TILE_K; kc += 32) {
      v16bf a, b;
      ((v4u*)&a)[0] = *(const v4u*)&As[abase + kc + 16 * half];
      ((v4u*)&a)[1] = *(const v4u*)&As[abase + kc + 16 * half + 8];
      ((v4u*)&b)[0] = *(const v4u*)&Bs[bbase + kc + 16 * half];
      ((v4u*)&b)[1] = *(const v4u*)&Bs[bbase + kc + 16 * half + 8];
      c = __builtin_amdgcn_wmma_f32_16x16x32_bf16(false, a, false, b, (short)0, c, false, false);
    }
    __syncthreads();
  }

  int ncol = n0 + wn * 16 + l15;
#pragma unroll
  for (int i = 0; i < 8; ++i) {
    int ml = wm * 16 + i + 8 * half;
    if (ml < rows)
      buf[(size_t)(base + ml) * HIDDEN + ncol] = c[i];
  }
}

// ---------------------------------------------------------------- combine
__global__ __launch_bounds__(256) void combine_kernel(
    const float* __restrict__ buf, const float* __restrict__ scores,
    const int* __restrict__ slot_of, float* __restrict__ out) {
  int s = blockIdx.x;
  float s0 = scores[2 * s], s1 = scores[2 * s + 1];
  const float4* b0 = (const float4*)(buf + (size_t)slot_of[2 * s] * HIDDEN);
  const float4* b1 = (const float4*)(buf + (size_t)slot_of[2 * s + 1] * HIDDEN);
  float4* o = (float4*)(out + (size_t)s * HIDDEN);
#pragma unroll
  for (int h = threadIdx.x; h < HIDDEN / 4; h += 256) {
    float4 a = b0[h], b = b1[h];
    float4 r;
    r.x = s0 * a.x + s1 * b.x;
    r.y = s0 * a.y + s1 * b.y;
    r.z = s0 * a.z + s1 * b.z;
    r.w = s0 * a.w + s1 * b.w;
    o[h] = r;
  }
}

// ---------------------------------------------------------------- launch
extern "C" void kernel_launch(void* const* d_in, const int* in_sizes, int n_in,
                              void* d_out, int out_size, void* d_ws, size_t ws_size,
                              hipStream_t stream) {
  const float* x     = (const float*)d_in[0];
  const float* wqkv  = (const float*)d_in[1];
  const float* w1    = (const float*)d_in[2];
  const float* w2    = (const float*)d_in[3];
  float* out         = (float*)d_out;

  char* ws = (char*)d_ws;
  size_t off = 0;
  unsigned short* inter = (unsigned short*)(ws + off); off += (size_t)NSLOTS * FFN * 2;    // 16.8 MB
  float* buf            = (float*)(ws + off);          off += (size_t)NSLOTS * HIDDEN * 4; // 33.6 MB
  float* scores         = (float*)(ws + off);          off += (size_t)NTOK * 2 * 4;
  int* indices          = (int*)(ws + off);            off += (size_t)NTOK * 2 * 4;
  int* list_token       = (int*)(ws + off);            off += (size_t)NSLOTS * 4;
  int* slot_of          = (int*)(ws + off);            off += (size_t)NSLOTS * 4;
  int* counts           = (int*)(ws + off);            // [0..7]=counts, [8..15]=cursors, [16..23]=offs
  int* cursors = counts + 8;
  int* offs    = counts + 16;

  zero_small_kernel<<<1, 32, 0, stream>>>(counts);
  router_kernel<<<NTOK / 8, 256, 0, stream>>>(x, wqkv, scores, indices, counts);
  scan_scatter_kernel<<<1, 256, 0, stream>>>(counts, cursors, offs, indices, list_token, slot_of);
  expert_fc1_kernel<<<dim3(FFN / TILE_N, NTOK / TILE_M, NE), 256, 0, stream>>>(
      x, w1, list_token, offs, counts, inter);
  expert_fc2_kernel<<<dim3(HIDDEN / TILE_N, NTOK / TILE_M, NE), 256, 0, stream>>>(
      inter, w2, offs, counts, buf);
  combine_kernel<<<NTOK, 256, 0, stream>>>(buf, scores, slot_of, out);
}
